// AttentionBlock_36893769073114
// MI455X (gfx1250) — matrix operand order repaired and checked by reading, not compiled, not run on hardware
//
#include <hip/hip_runtime.h>
#include <hip/hip_bf16.h>

typedef __attribute__((ext_vector_type(16))) _Float16 v16h;
typedef __attribute__((ext_vector_type(8)))  _Float16 v8h;
typedef __attribute__((ext_vector_type(8)))  float    v8f;

#define BATCH 8
#define CCH   512
#define LSP   1024
#define HEADS 8
#define DH    64
#define NGRP  32
#define CPG   16          // CCH / NGRP
#define GN_EPS 1e-5f
#define ATT_SCALE 0.125f  // dh^-0.5

#define APAD 40   // padded row length (halves) for A tiles   (80B rows, 16B aligned)
#define BPAD 40   // padded row length (halves) for B^T tiles (80B rows)
#define KPAD 72   // padded row length for 64-wide attention tiles (144B rows)

// ---------------------------------------------------------------------------
// WMMA fragment helpers (wave32, v_wmma_f32_16x16x32_f16)
// A (16x32 f16): m = lane&15; halves 0..7 -> k = 8*laneHi + i, 8..15 -> +16
// B (32x16 f16): n = lane&15; halves 0..7 -> k = 8*laneHi + i, 8..15 -> +16
// C/D (16x16 f32): m = v + 8*laneHi, n = lane&15
// ---------------------------------------------------------------------------
__device__ __forceinline__ v16h load16(const _Float16* p) {
  union { v16h v; v8h h[2]; } u;
  u.h[0] = *(const v8h*)p;
  u.h[1] = *(const v8h*)(p + 8);
  return u.v;
}
__device__ __forceinline__ void store16(_Float16* p, v16h x) {
  union { v16h v; v8h h[2]; } u;
  u.v = x;
  *(v8h*)p       = u.h[0];
  *(v8h*)(p + 8) = u.h[1];
}
// A fragment from row-major [m][k] tile with row pitch lda (halves)
__device__ __forceinline__ v16h load_A16(const _Float16* p, int lda,
                                         int laneN, int laneHi) {
  const _Float16* r = p + laneN * lda + laneHi * 8;
  union { v16h v; v8h h[2]; } u;
  u.h[0] = *(const v8h*)r;
  u.h[1] = *(const v8h*)(r + 16);
  return u.v;
}
// B fragment from TRANSPOSED [n][k] tile with row pitch ldp (halves).
__device__ __forceinline__ v16h load_Bt(const _Float16* t, int ldp, int n0,
                                        int k0, int laneN, int laneHi) {
  const _Float16* r = t + (size_t)(n0 + laneN) * ldp + k0 + laneHi * 8;
  union { v16h v; v8h h[2]; } u;
  u.h[0] = *(const v8h*)r;
  u.h[1] = *(const v8h*)(r + 16);
  return u.v;
}

// CDNA5 async global->LDS copy of 32 contiguous bytes (2x B128), ASYNCcnt.
// INST_OFFSET is added to both the LDS and global addresses (ISA 15.18.3).
__device__ __forceinline__ void async_copy32B(_Float16* lds,
                                              const _Float16* g) {
  unsigned l32 = (unsigned)(unsigned long long)(uintptr_t)lds;
  unsigned long long g64 = (unsigned long long)(uintptr_t)g;
  asm volatile(
      "global_load_async_to_lds_b128 %0, %1, off\n\t"
      "global_load_async_to_lds_b128 %0, %1, off offset:16"
      :
      : "v"(l32), "v"(g64)
      : "memory");
}
__device__ __forceinline__ void wait_async0() {
  asm volatile("s_wait_asynccnt 0x0" ::: "memory");
}

// ---------------------------------------------------------------------------
// Kernel 1: GroupNorm  x(b,c,L) fp32 -> hnT(b,L,C) f16  (l-major!)
// one block per (b, group): 16 channels x 1024 positions
// ---------------------------------------------------------------------------
__global__ __launch_bounds__(256) void gn_kernel(
    const float* __restrict__ x, const float* __restrict__ gamma,
    const float* __restrict__ beta, _Float16* __restrict__ hnT) {
  const int bg = blockIdx.x;
  const int b = bg / NGRP, g = bg % NGRP;
  const int tid = threadIdx.x, w = tid >> 5, lane = tid & 31;
  const float* xg = x + ((size_t)b * CCH + (size_t)g * CPG) * LSP;

  float s = 0.f, s2 = 0.f;
#pragma unroll
  for (int c = 0; c < CPG; c++) {
#pragma unroll
    for (int lc = 0; lc < 4; lc++) {
      float v = xg[(size_t)c * LSP + lc * 256 + tid];
      s += v;
      s2 += v * v;
    }
  }
#pragma unroll
  for (int m = 16; m >= 1; m >>= 1) {
    s  += __shfl_xor(s,  m, 32);
    s2 += __shfl_xor(s2, m, 32);
  }
  __shared__ float rs[8], rs2[8];
  if (lane == 0) { rs[w] = s; rs2[w] = s2; }
  __syncthreads();
  float S = 0.f, S2 = 0.f;
#pragma unroll
  for (int i = 0; i < 8; i++) { S += rs[i]; S2 += rs2[i]; }
  const float cntInv = 1.f / (float)(CPG * LSP);
  const float mean = S * cntInv;
  const float var  = S2 * cntInv - mean * mean;
  const float inv  = rsqrtf(var + GN_EPS);

  float gm[CPG], bt[CPG];
#pragma unroll
  for (int c = 0; c < CPG; c++) {
    gm[c] = gamma[g * CPG + c] * inv;
    bt[c] = beta[g * CPG + c];
  }
  _Float16* hb = hnT + (size_t)b * LSP * CCH + g * CPG;
#pragma unroll
  for (int lc = 0; lc < 4; lc++) {
    int l = lc * 256 + tid;
    v16h t;
#pragma unroll
    for (int c = 0; c < CPG; c++)
      t[c] = (_Float16)((xg[(size_t)c * LSP + l] - mean) * gm[c] + bt[c]);
    store16(hb + (size_t)l * CCH, t);
  }
}

// ---------------------------------------------------------------------------
// Kernel 2: fused QKV GEMM per batch (M=1536, N=1024, K=512).
//   rows 0..511    : q -> qT (b,h,L,dh)    (+bq)*scale
//   rows 512..1023 : k -> kT (b,h,L,dh)    (+bkv)*scale
//   rows 1024..1535: v -> vM (b,h*dh,L)    (+bkv)
// ---------------------------------------------------------------------------
__global__ __launch_bounds__(256) void qkv_gemm_kernel(
    const float* __restrict__ Wq,  const float* __restrict__ bq,
    const float* __restrict__ Wkv, const float* __restrict__ bkv,
    const _Float16* __restrict__ hnT,
    _Float16* __restrict__ qT, _Float16* __restrict__ kT,
    _Float16* __restrict__ vM) {
  __shared__ __align__(16) _Float16 aLds[128][APAD];  // [m][k]
  __shared__ __align__(16) _Float16 bT[128][BPAD];    // [n][k] transposed
  const int b = blockIdx.z;
  const int n0 = blockIdx.x * 128;
  const int m0 = blockIdx.y * 128;
  const int tid = threadIdx.x, w = tid >> 5, lane = tid & 31;
  const int wtm = w >> 2, wtn = w & 3;  // 2 x 4 wave grid
  const int laneN = lane & 15, laneHi = lane >> 4;

  v8f acc[4][2];
  v8f zero = {};
#pragma unroll
  for (int i = 0; i < 4; i++)
#pragma unroll
    for (int j = 0; j < 2; j++) acc[i][j] = zero;

  const _Float16* hbT = hnT + (size_t)b * LSP * CCH;

  const int aRow = tid >> 1, aKK = (tid & 1) * 16;  // A: 128x32 fp32->f16
  const int oRow = m0 + aRow;
  const float* wrow =
      (oRow < 512) ? (Wq + (size_t)oRow * CCH) : (Wkv + (size_t)(oRow - 512) * CCH);
  const int bL = tid >> 1, bCQ = (tid & 1) * 16;    // B^T: contiguous rows of hnT

  for (int k0 = 0; k0 < CCH; k0 += 32) {
    {  // stage A (convert fp32 -> f16)
      v16h t;
#pragma unroll
      for (int i = 0; i < 16; i++) t[i] = (_Float16)wrow[k0 + aKK + i];
      store16(&aLds[aRow][aKK], t);
    }
    // stage B^T: async contiguous copy from l-major hnT
    async_copy32B(&bT[bL][bCQ], hbT + (size_t)(n0 + bL) * CCH + k0 + bCQ);
    wait_async0();
    __syncthreads();

    v16h af[4], bf[2];
#pragma unroll
    for (int i = 0; i < 4; i++)
      af[i] = load_A16(&aLds[wtm * 64 + i * 16][0], APAD, laneN, laneHi);
#pragma unroll
    for (int j = 0; j < 2; j++)
      bf[j] = load_Bt(&bT[0][0], BPAD, wtn * 32 + j * 16, 0, laneN, laneHi);
#pragma unroll
    for (int i = 0; i < 4; i++)
#pragma unroll
      for (int j = 0; j < 2; j++)
        acc[i][j] = __builtin_amdgcn_wmma_f32_16x16x32_f16(
            false, af[i], false, bf[j], (short)0, acc[i][j], false, false);
    __syncthreads();
  }

#pragma unroll
  for (int i = 0; i < 4; i++) {
#pragma unroll
    for (int j = 0; j < 2; j++) {
#pragma unroll
      for (int v = 0; v < 8; v++) {
        int o = m0 + wtm * 64 + i * 16 + laneHi * 8 + v;
        int l = n0 + wtn * 32 + j * 16 + laneN;
        float val = acc[i][j][v];
        if (o < 512) {
          int h = o >> 6, d = o & 63;
          qT[(((size_t)(b * HEADS + h)) * LSP + l) * DH + d] =
              (_Float16)((val + bq[o]) * ATT_SCALE);
        } else if (o < 1024) {
          int kk = o - 512;
          int h = kk >> 6, d = kk & 63;
          kT[(((size_t)(b * HEADS + h)) * LSP + l) * DH + d] =
              (_Float16)((val + bkv[kk]) * ATT_SCALE);
        } else {
          int vv = o - 1024;
          vM[((size_t)(b * CCH + vv)) * LSP + l] =
              (_Float16)(val + bkv[vv + 512]);
        }
      }
    }
  }
}

// ---------------------------------------------------------------------------
// Kernel 3: flash attention per (b,h). 8 waves x 16 query rows = 128 rows
// per block; async-stage K/V 64-column blocks; online softmax.
// Output -> aH (b, L, inner) f16 (l-major) for the projection GEMM.
// ---------------------------------------------------------------------------
__global__ __launch_bounds__(256) void attn_kernel(
    const _Float16* __restrict__ qT, const _Float16* __restrict__ kT,
    const _Float16* __restrict__ vM, _Float16* __restrict__ aH) {
  __shared__ __align__(16) _Float16 kTl[64][KPAD];     // [s][d]  (B^T for QK)
  __shared__ __align__(16) _Float16 vTl[64][KPAD];     // [d][s]  (B^T for PV)
  __shared__ __align__(16) _Float16 pLds[8][16][KPAD]; // wave-private P
  const int bh = blockIdx.y;
  const int bI = bh >> 3, hI = bh & 7;
  const int tid = threadIdx.x, w = tid >> 5, lane = tid & 31;
  const int laneN = lane & 15, laneHi = lane >> 4;
  const int t0 = blockIdx.x * 128 + w * 16;

  const _Float16* qbase = qT + ((size_t)bh * LSP + t0) * DH;
  v16h qa[2];
  qa[0] = load_A16(qbase,      DH, laneN, laneHi);
  qa[1] = load_A16(qbase + 32, DH, laneN, laneHi);

  v8f accO[4];
  v8f zero = {};
#pragma unroll
  for (int jd = 0; jd < 4; jd++) accO[jd] = zero;
  float run_m[8], run_l[8];
#pragma unroll
  for (int v = 0; v < 8; v++) { run_m[v] = -1e30f; run_l[v] = 0.f; }

  const _Float16* kbase = kT + (size_t)bh * LSP * DH;             // (s, d) rows
  const _Float16* vbase = vM + (size_t)(bI * CCH + hI * DH) * LSP;// (d, s) rows

  const int stR = tid >> 2, stQ = (tid & 3) * 16;  // staging role

  for (int s0 = 0; s0 < LSP; s0 += 64) {
    // async stage K^T [s][d] and V^T [d][s] (both contiguous in memory)
    async_copy32B(&kTl[stR][stQ], kbase + (size_t)(s0 + stR) * DH + stQ);
    async_copy32B(&vTl[stR][stQ], vbase + (size_t)stR * LSP + s0 + stQ);
    wait_async0();
    __syncthreads();

    // S = Q K^T (16 x 64), k-dim = dh = 64 -> 2 frags
    v8f sacc[4];
#pragma unroll
    for (int j = 0; j < 4; j++) sacc[j] = zero;
#pragma unroll
    for (int kf = 0; kf < 2; kf++) {
#pragma unroll
      for (int j = 0; j < 4; j++) {
        v16h kb = load_Bt(&kTl[0][0], KPAD, j * 16, kf * 32, laneN, laneHi);
        sacc[j] = __builtin_amdgcn_wmma_f32_16x16x32_f16(
            false, qa[kf], false, kb, (short)0, sacc[j], false, false);
      }
    }

    // online softmax over this 64-column block
    float pvals[4][8];
#pragma unroll
    for (int v = 0; v < 8; v++) {
      float tm = sacc[0][v];
#pragma unroll
      for (int j = 1; j < 4; j++) tm = fmaxf(tm, sacc[j][v]);
#pragma unroll
      for (int m = 1; m < 16; m <<= 1) tm = fmaxf(tm, __shfl_xor(tm, m, 32));
      float nm   = fmaxf(run_m[v], tm);
      float corr = __expf(run_m[v] - nm);
      float ts = 0.f;
#pragma unroll
      for (int j = 0; j < 4; j++) {
        float p = __expf(sacc[j][v] - nm);
        pvals[j][v] = p;
        ts += p;
      }
#pragma unroll
      for (int m = 1; m < 16; m <<= 1) ts += __shfl_xor(ts, m, 32);
      run_l[v] = run_l[v] * corr + ts;
      run_m[v] = nm;
#pragma unroll
      for (int jd = 0; jd < 4; jd++) accO[jd][v] *= corr;
    }

    // D-fragment -> A-fragment via wave-private LDS
#pragma unroll
    for (int j = 0; j < 4; j++)
#pragma unroll
      for (int v = 0; v < 8; v++)
        pLds[w][laneHi * 8 + v][j * 16 + laneN] = (_Float16)pvals[j][v];

    // O += P V^T  (k-dim = 64 s-values -> 2 frags)
#pragma unroll
    for (int kf = 0; kf < 2; kf++) {
      v16h pa = load_A16(&pLds[w][0][kf * 32], KPAD, laneN, laneHi);
#pragma unroll
      for (int jd = 0; jd < 4; jd++) {
        v16h vb = load_Bt(&vTl[0][0], KPAD, jd * 16, kf * 32, laneN, laneHi);
        accO[jd] = __builtin_amdgcn_wmma_f32_16x16x32_f16(
            false, pa, false, vb, (short)0, accO[jd], false, false);
      }
    }
    __syncthreads();
  }

  float invl[8];
#pragma unroll
  for (int v = 0; v < 8; v++) invl[v] = 1.f / run_l[v];
  // store l-major: aH[b][t][h*dh + d]
  _Float16* ob = aH + (size_t)bI * LSP * CCH + hI * DH;
#pragma unroll
  for (int jd = 0; jd < 4; jd++) {
#pragma unroll
    for (int v = 0; v < 8; v++) {
      int d = jd * 16 + laneN;
      int t = t0 + laneHi * 8 + v;
      ob[(size_t)t * CCH + d] = (_Float16)(accO[jd][v] * invl[v]);
    }
  }
}

// ---------------------------------------------------------------------------
// Kernel 4: output projection + bias + residual, fp32 out.
// out(b,c,L) = x + Wo @ aH + bo ;  M=512, N=1024, K=512, aH is (b,L,K)
// ---------------------------------------------------------------------------
__global__ __launch_bounds__(256) void out_gemm_kernel(
    const float* __restrict__ Wo, const float* __restrict__ bo,
    const _Float16* __restrict__ aH, const float* __restrict__ xin,
    float* __restrict__ out) {
  __shared__ __align__(16) _Float16 aLds[128][APAD];
  __shared__ __align__(16) _Float16 bT[128][BPAD];
  const int b = blockIdx.z;
  const int n0 = blockIdx.x * 128;
  const int m0 = blockIdx.y * 128;
  const int tid = threadIdx.x, w = tid >> 5, lane = tid & 31;
  const int wtm = w >> 2, wtn = w & 3;
  const int laneN = lane & 15, laneHi = lane >> 4;

  v8f acc[4][2];
  v8f zero = {};
#pragma unroll
  for (int i = 0; i < 4; i++)
#pragma unroll
    for (int j = 0; j < 2; j++) acc[i][j] = zero;

  const _Float16* hbT = aH + (size_t)b * LSP * CCH;  // (l, i)
  const int aRow = tid >> 1, aKK = (tid & 1) * 16;
  const float* wrow = Wo + (size_t)(m0 + aRow) * CCH;
  const int bL = tid >> 1, bCQ = (tid & 1) * 16;

  for (int k0 = 0; k0 < CCH; k0 += 32) {
    {
      v16h t;
#pragma unroll
      for (int i = 0; i < 16; i++) t[i] = (_Float16)wrow[k0 + aKK + i];
      store16(&aLds[aRow][aKK], t);
    }
    async_copy32B(&bT[bL][bCQ], hbT + (size_t)(n0 + bL) * CCH + k0 + bCQ);
    wait_async0();
    __syncthreads();

    v16h af[4], bf[2];
#pragma unroll
    for (int i = 0; i < 4; i++)
      af[i] = load_A16(&aLds[wtm * 64 + i * 16][0], APAD, laneN, laneHi);
#pragma unroll
    for (int j = 0; j < 2; j++)
      bf[j] = load_Bt(&bT[0][0], BPAD, wtn * 32 + j * 16, 0, laneN, laneHi);
#pragma unroll
    for (int i = 0; i < 4; i++)
#pragma unroll
      for (int j = 0; j < 2; j++)
        acc[i][j] = __builtin_amdgcn_wmma_f32_16x16x32_f16(
            false, af[i], false, bf[j], (short)0, acc[i][j], false, false);
    __syncthreads();
  }

#pragma unroll
  for (int i = 0; i < 4; i++) {
#pragma unroll
    for (int j = 0; j < 2; j++) {
#pragma unroll
      for (int v = 0; v < 8; v++) {
        int o = m0 + wtm * 64 + i * 16 + laneHi * 8 + v;
        int l = n0 + wtn * 32 + j * 16 + laneN;
        size_t idx = ((size_t)b * CCH + o) * LSP + l;
        out[idx] = acc[i][j][v] + bo[o] + xin[idx];
      }
    }
  }
}

// ---------------------------------------------------------------------------
// launch
// ---------------------------------------------------------------------------
extern "C" void kernel_launch(void* const* d_in, const int* in_sizes, int n_in,
                              void* d_out, int out_size, void* d_ws,
                              size_t ws_size, hipStream_t stream) {
  (void)in_sizes; (void)n_in; (void)out_size; (void)ws_size;
  const float* x     = (const float*)d_in[0];
  const float* gamma = (const float*)d_in[1];
  const float* beta  = (const float*)d_in[2];
  const float* Wq    = (const float*)d_in[3];
  const float* bq    = (const float*)d_in[4];
  const float* Wkv   = (const float*)d_in[5];
  const float* bkv   = (const float*)d_in[6];
  const float* Wo    = (const float*)d_in[7];
  const float* bo    = (const float*)d_in[8];
  float* out = (float*)d_out;

  // workspace: 5 f16 segments of b*c*L = 4,194,304 elems -> 40 MiB total
  _Float16* ws = (_Float16*)d_ws;
  const size_t SEG = (size_t)BATCH * CCH * LSP;
  _Float16* hnT = ws;            // (b, L, C)
  _Float16* qT  = ws + 1 * SEG;  // (b, h, L, dh)
  _Float16* kT  = ws + 2 * SEG;  // (b, h, L, dh)
  _Float16* vM  = ws + 3 * SEG;  // (b, h*dh, L)
  _Float16* aH  = ws + 4 * SEG;  // (b, L, inner)

  gn_kernel<<<dim3(BATCH * NGRP), 256, 0, stream>>>(x, gamma, beta, hnT);
  qkv_gemm_kernel<<<dim3(LSP / 128, 1536 / 128, BATCH), 256, 0, stream>>>(
      Wq, bq, Wkv, bkv, hnT, qT, kT, vM);
  attn_kernel<<<dim3(LSP / 128, BATCH * HEADS), 256, 0, stream>>>(qT, kT, vM,
                                                                  aH);
  out_gemm_kernel<<<dim3(LSP / 128, CCH / 128, BATCH), 256, 0, stream>>>(
      Wo, bo, aH, x, out);
}
